// StreamingMamba2_58119497450424
// MI455X (gfx1250) — matrix operand order repaired and checked by reading, not compiled, not run on hardware
//
#include <hip/hip_runtime.h>
#include <hip/hip_bf16.h>

typedef __attribute__((ext_vector_type(16))) __bf16 v16bf;
typedef __attribute__((ext_vector_type(8)))  float  v8f;

#define DEV __device__ __forceinline__

static constexpr int BB    = 2;
static constexpr int L     = 3136;       // 56*56
static constexpr int DM    = 192;        // d_model
static constexpr int DIN   = 384;        // d_inner
static constexpr int NH    = 6;          // heads
static constexpr int HD    = 64;         // headdim == d_state
static constexpr int CONVD = 512;        // conv_dim
static constexpr int NPROJ = 902;        // d_in_proj
static constexpr int NPAD  = 928;        // 29*32
static constexpr int M     = BB * L;     // 6272 rows

// ---------------- WMMA helpers ----------------
DEV v8f vz() { v8f z = {0.f,0.f,0.f,0.f,0.f,0.f,0.f,0.f}; return z; }

DEV v8f wmma_bf(v16bf a, v16bf b, v8f c) {
  return __builtin_amdgcn_wmma_f32_16x16x32_bf16(false, a, false, b, (short)0, c, false, false);
}

// A-operand (16x32, M x K): lane<16: m=lane, K {kb..kb+7, kb+16..kb+23}, kb=0; lane>=16: kb=8
DEV v16bf ldfragA(const __hip_bfloat16* p, int ld) {
  int lane = threadIdx.x & 31;
  const __hip_bfloat16* q = p + (size_t)(lane & 15) * ld + ((lane >> 4) << 3);
  union { v16bf v; uint4 u[2]; } t;
  t.u[0] = *(const uint4*)(q);
  t.u[1] = *(const uint4*)(q + 16);
  return t.v;
}
// B-operand (32x16, K x N) loaded from an N x K row-major view:
// lane<16: n=lane, K {kb..kb+7, kb+16..kb+23}, kb=0; lane>=16: kb=8
DEV v16bf ldfragB(const __hip_bfloat16* p, int ld) {
  int lane = threadIdx.x & 31;
  const __hip_bfloat16* q = p + (size_t)(lane & 15) * ld + ((lane >> 4) << 3);
  union { v16bf v; uint4 u[2]; } t;
  t.u[0] = *(const uint4*)(q);
  t.u[1] = *(const uint4*)(q + 16);
  return t.v;
}
// C/D tile (16x16 f32): n = n0 + lane%16 ; m = m0 + (lane/16)*8 + i
DEV void storeC(float* C, int m0, int n0, int ldc, v8f acc) {
  int lane = threadIdx.x & 31;
  int n  = n0 + (lane & 15);
  int mb = m0 + ((lane >> 4) << 3);
#pragma unroll
  for (int i = 0; i < 8; i++) C[(size_t)(mb + i) * ldc + n] = acc[i];
}

// ---------------- repack kernels ----------------
__global__ void k_u2bf(const float* __restrict__ u, __hip_bfloat16* __restrict__ ubf) {
  size_t idx = (size_t)blockIdx.x * 256 + threadIdx.x;   // < M*DM
  if (idx >= (size_t)M * DM) return;
  int c = idx % DM; size_t row = idx / DM;
  int b = row / L, l = row % L;
  ubf[idx] = __float2bfloat16(u[((size_t)b * DM + c) * L + l]);
}
__global__ void k_win2bf(const float* __restrict__ w, __hip_bfloat16* __restrict__ wbf) {
  size_t idx = (size_t)blockIdx.x * 256 + threadIdx.x;   // < NPAD*DM
  if (idx >= (size_t)NPAD * DM) return;
  int k = idx % DM; int n = idx / DM;
  wbf[idx] = __float2bfloat16((n < NPROJ) ? w[(size_t)n * DM + k] : 0.f);
}
__global__ void k_wout2bf(const float* __restrict__ w, __hip_bfloat16* __restrict__ wbf) {
  size_t idx = (size_t)blockIdx.x * 256 + threadIdx.x;   // < DM*DIN
  if (idx >= (size_t)DM * DIN) return;
  wbf[idx] = __float2bfloat16(w[idx]);
}

// ---------------- in-proj GEMM: zx(M x NPAD) = ubf(M x 192) * winbf(NPAD x 192)^T
__global__ __launch_bounds__(256) void k_gemm_in(const __hip_bfloat16* __restrict__ A,
                                                 const __hip_bfloat16* __restrict__ W,
                                                 float* __restrict__ Cm) {
  int wave = threadIdx.x >> 5;
  int tile = blockIdx.x * 8 + wave;          // 196 * 29 tiles of 32x32
  if (tile >= 196 * 29) return;
  int mt = tile / 29, nt = tile % 29;
  int m0 = mt * 32, n0 = nt * 32;
  v8f a00 = vz(), a01 = vz(), a10 = vz(), a11 = vz();
  for (int k0 = 0; k0 < DM; k0 += 32) {
    v16bf fa0 = ldfragA(A + (size_t)m0 * DM + k0, DM);
    v16bf fa1 = ldfragA(A + (size_t)(m0 + 16) * DM + k0, DM);
    v16bf fb0 = ldfragB(W + (size_t)n0 * DM + k0, DM);
    v16bf fb1 = ldfragB(W + (size_t)(n0 + 16) * DM + k0, DM);
    a00 = wmma_bf(fa0, fb0, a00);
    a01 = wmma_bf(fa0, fb1, a01);
    a10 = wmma_bf(fa1, fb0, a10);
    a11 = wmma_bf(fa1, fb1, a11);
  }
  storeC(Cm, m0,      n0,      NPAD, a00);
  storeC(Cm, m0,      n0 + 16, NPAD, a01);
  storeC(Cm, m0 + 16, n0,      NPAD, a10);
  storeC(Cm, m0 + 16, n0 + 16, NPAD, a11);
}

// ---------------- dt = softplus(dt_raw + bias) ----------------
__global__ void k_dt(const float* __restrict__ zx, const float* __restrict__ dt_bias,
                     float* __restrict__ dtv) {
  size_t idx = (size_t)blockIdx.x * 256 + threadIdx.x;   // < M*NH
  if (idx >= (size_t)M * NH) return;
  int h = idx % NH; size_t row = idx / NH;
  float v = zx[row * NPAD + (DIN + CONVD) + h] + dt_bias[h];
  dtv[idx] = (v > 20.f) ? v : log1pf(__expf(v));
}

// ---------------- depthwise 3x3 conv + SiLU ----------------
__global__ void k_conv(const float* __restrict__ zx, const float* __restrict__ cw,
                       const float* __restrict__ cb, float* __restrict__ xbcc) {
  size_t idx = (size_t)blockIdx.x * 256 + threadIdx.x;   // < M*CONVD
  if (idx >= (size_t)M * CONVD) return;
  int ch = idx % CONVD; size_t row = idx / CONVD;
  int b = row / L, l = row % L;
  int y = l / 56, x = l % 56;
  float acc = cb[ch];
#pragma unroll
  for (int ky = 0; ky < 3; ky++) {
    int yy = y + ky - 1;
    if (yy < 0 || yy >= 56) continue;
#pragma unroll
    for (int kx = 0; kx < 3; kx++) {
      int xx = x + kx - 1;
      if (xx < 0 || xx >= 56) continue;
      acc += zx[((size_t)b * L + yy * 56 + xx) * NPAD + DIN + ch] * cw[(ky * 3 + kx) * CONVD + ch];
    }
  }
  xbcc[idx] = acc / (1.f + __expf(-acc));   // silu
}

// ---------------- cumsum of A*dt per (b,h) ----------------
__global__ void k_cumsum(const float* __restrict__ dtv, const float* __restrict__ A_log,
                         float* __restrict__ dcum) {
  int bh = blockIdx.x; int b = bh / NH; int h = bh % NH;
  float Ac = -__expf(A_log[h]);
  __shared__ float buf[256];
  __shared__ float carry;
  int tid = threadIdx.x;
  if (tid == 0) carry = 0.f;
  __syncthreads();
  for (int base = 0; base < L; base += 256) {
    int i = base + tid;
    float v = (i < L) ? Ac * dtv[((size_t)b * L + i) * NH + h] : 0.f;
    buf[tid] = v; __syncthreads();
    for (int off = 1; off < 256; off <<= 1) {
      float t = (tid >= off) ? buf[tid - off] : 0.f;
      __syncthreads();
      buf[tid] += t;
      __syncthreads();
    }
    float res = buf[tid] + carry;
    if (i < L) dcum[(size_t)bh * L + i] = res;
    __syncthreads();
    if (tid == 0) carry += buf[255];
    __syncthreads();
  }
}

// ---------------- bf16 operand prep for SSD ----------------
__global__ void k_prep(const float* __restrict__ xbcc, const float* __restrict__ dtv,
                       __hip_bfloat16* __restrict__ xtbf, __hip_bfloat16* __restrict__ bbf,
                       __hip_bfloat16* __restrict__ cbf) {
  size_t idx = (size_t)blockIdx.x * 256 + threadIdx.x;   // < M*CONVD
  if (idx >= (size_t)M * CONVD) return;
  int ch = idx % CONVD; size_t row = idx / CONVD;
  int b = row / L, l = row % L;
  float v = xbcc[idx];
  if (ch < DIN) {
    int h = ch >> 6, p = ch & 63;
    float xv = v * dtv[row * NH + h];
    xtbf[((size_t)(b * NH + h) * HD + p) * L + l] = __float2bfloat16(xv);   // transposed (p, l)
  } else if (ch < DIN + HD) {
    bbf[row * HD + (ch - DIN)] = __float2bfloat16(v);
  } else {
    cbf[row * HD + (ch - DIN - HD)] = __float2bfloat16(v);
  }
}

// ---------------- causal quadratic SSD: Y_diag ----------------
__global__ __launch_bounds__(128) void k_attn(const __hip_bfloat16* __restrict__ cbf,
                                              const __hip_bfloat16* __restrict__ bbf,
                                              const __hip_bfloat16* __restrict__ xtbf,
                                              const float* __restrict__ dcum,
                                              float* __restrict__ Y) {
  __shared__ alignas(16) __hip_bfloat16 sS[4][16 * 32];
  int wave = threadIdx.x >> 5, lane = threadIdx.x & 31;
  int bh = blockIdx.y; int b = bh / NH; int h = bh % NH;
  int r0 = blockIdx.x * 64 + wave * 16;
  const __hip_bfloat16* cb  = cbf + (size_t)b * L * HD;
  const __hip_bfloat16* bbp = bbf + (size_t)b * L * HD;
  const __hip_bfloat16* xt  = xtbf + (size_t)bh * HD * L;
  const float* dc = dcum + (size_t)bh * L;

  v16bf ca0 = ldfragA(cb + (size_t)r0 * HD + 0, HD);
  v16bf ca1 = ldfragA(cb + (size_t)r0 * HD + 32, HD);
  int mb = r0 + ((lane >> 4) << 3);
  float dl[8];
#pragma unroll
  for (int i = 0; i < 8; i++) dl[i] = dc[mb + i];

  v8f y0 = vz(), y1 = vz(), y2 = vz(), y3 = vz();
  __hip_bfloat16* sbuf = sS[wave];

  for (int s0 = 0; s0 < r0 + 16; s0 += 32) {
    // S tile (16 x 32) = C(16 x 64) * B^T(64 x 32)
    v16bf b0a = ldfragB(bbp + (size_t)s0 * HD + 0, HD);
    v16bf b0b = ldfragB(bbp + (size_t)s0 * HD + 32, HD);
    v16bf b1a = ldfragB(bbp + (size_t)(s0 + 16) * HD + 0, HD);
    v16bf b1b = ldfragB(bbp + (size_t)(s0 + 16) * HD + 32, HD);
    v8f s0a = vz(), s1a = vz();
    s0a = wmma_bf(ca0, b0a, s0a); s0a = wmma_bf(ca1, b0b, s0a);
    s1a = wmma_bf(ca0, b1a, s1a); s1a = wmma_bf(ca1, b1b, s1a);
    // decay + causal mask (exp arg <= 0, safe)
    int sa = s0 + (lane & 15), sb = sa + 16;
    float dsa = dc[sa], dsb = dc[sb];
#pragma unroll
    for (int i = 0; i < 8; i++) {
      int mrow = mb + i;
      float f0 = (sa <= mrow) ? __expf(dl[i] - dsa) : 0.f;
      float f1 = (sb <= mrow) ? __expf(dl[i] - dsb) : 0.f;
      s0a[i] *= f0; s1a[i] *= f1;
    }
    // S -> bf16 via per-wave LDS roundtrip (C-layout -> A-fragment layout)
#pragma unroll
    for (int i = 0; i < 8; i++) {
      int rr = ((lane >> 4) << 3) + i;
      sbuf[rr * 32 + (lane & 15)]      = __float2bfloat16(s0a[i]);
      sbuf[rr * 32 + 16 + (lane & 15)] = __float2bfloat16(s1a[i]);
    }
    __builtin_amdgcn_wave_barrier();   // per-wave LDS in-order; block compiler reordering
    v16bf sfrag = ldfragA(sbuf, 32);
    __builtin_amdgcn_wave_barrier();
    // Y(16 x 64) += S(16 x 32) * X(32 x 64)  (X transposed layout: rows p, ld L)
    y0 = wmma_bf(sfrag, ldfragB(xt + (size_t)0  * L + s0, L), y0);
    y1 = wmma_bf(sfrag, ldfragB(xt + (size_t)16 * L + s0, L), y1);
    y2 = wmma_bf(sfrag, ldfragB(xt + (size_t)32 * L + s0, L), y2);
    y3 = wmma_bf(sfrag, ldfragB(xt + (size_t)48 * L + s0, L), y3);
  }
  int nl = lane & 15;
#pragma unroll
  for (int i = 0; i < 8; i++) {
    size_t base = ((size_t)b * L + mb + i) * DIN + h * HD + nl;
    Y[base +  0] = y0[i];
    Y[base + 16] = y1[i];
    Y[base + 32] = y2[i];
    Y[base + 48] = y3[i];
  }
}

// ---------------- inter-chunk states + final_state ----------------
__global__ void k_states(const __hip_bfloat16* __restrict__ bbf,
                         const __hip_bfloat16* __restrict__ xtbf,
                         const float* __restrict__ dcum, const float* __restrict__ state_in,
                         float* __restrict__ fs_out) {
  int bh = blockIdx.x; int b = bh / NH;
  int tid = threadIdx.x;
  int p = tid >> 2;            // 0..63
  int n0 = (tid & 3) * 16;
  const __hip_bfloat16* xt = xtbf + ((size_t)bh * HD + p) * L;
  const __hip_bfloat16* bb = bbf + (size_t)b * L * HD;
  const float* dc = dcum + (size_t)bh * L;
  float dlast = dc[L - 1];
  float acc[16];
#pragma unroll
  for (int j = 0; j < 16; j++) acc[j] = 0.f;
  for (int l = 0; l < L; l++) {
    float xw = __bfloat162float(xt[l]) * __expf(dlast - dc[l]);
    const __hip_bfloat16* br = bb + (size_t)l * HD + n0;
#pragma unroll
    for (int j = 0; j < 16; j++) acc[j] += xw * __bfloat162float(br[j]);
  }
  float edl = __expf(dlast);
  const float* st = state_in + (size_t)bh * (HD * HD) + p * HD + n0;
  float* out = fs_out + (size_t)bh * (HD * HD) + p * HD + n0;
#pragma unroll
  for (int j = 0; j < 16; j++) out[j] = edl * st[j] + acc[j];
}

// ---------------- Y_off += exp(cumA) * (C . state_prev) ----------------
__global__ void k_yoff(const __hip_bfloat16* __restrict__ cbf, const float* __restrict__ state_in,
                       const float* __restrict__ dcum, float* __restrict__ Y) {
  size_t idx = (size_t)blockIdx.x * 256 + threadIdx.x;   // < M*NH*HD
  if (idx >= (size_t)M * NH * HD) return;
  int p = idx & 63; size_t t = idx >> 6;
  int h = t % NH; size_t row = t / NH;
  int b = row / L, l = row % L;
  const float* st = state_in + ((size_t)b * NH + h) * (HD * HD) + p * HD;
  const __hip_bfloat16* cr = cbf + row * HD;
  float acc = 0.f;
#pragma unroll 8
  for (int n = 0; n < HD; n++) acc += __bfloat162float(cr[n]) * st[n];
  Y[row * DIN + h * HD + p] += __expf(dcum[((size_t)b * NH + h) * L + l]) * acc;
}

// ---------------- layernorm + z-gate -> bf16 ----------------
__global__ void k_ln(const float* __restrict__ Y, const float* __restrict__ zx,
                     const float* __restrict__ gamma, const float* __restrict__ beta,
                     __hip_bfloat16* __restrict__ ybf) {
  int wave = threadIdx.x >> 5, lane = threadIdx.x & 31;
  size_t row = (size_t)blockIdx.x * 8 + wave;
  float vals[12];
  float s = 0.f;
#pragma unroll
  for (int j = 0; j < 12; j++) { float v = Y[row * DIN + lane + j * 32]; vals[j] = v; s += v; }
#pragma unroll
  for (int o = 16; o >= 1; o >>= 1) s += __shfl_xor(s, o, 32);
  float mu = s * (1.f / DIN);
  float s2 = 0.f;
#pragma unroll
  for (int j = 0; j < 12; j++) { float d = vals[j] - mu; s2 += d * d; }
#pragma unroll
  for (int o = 16; o >= 1; o >>= 1) s2 += __shfl_xor(s2, o, 32);
  float rstd = rsqrtf(s2 * (1.f / DIN) + 1e-5f);
#pragma unroll
  for (int j = 0; j < 12; j++) {
    int c = lane + j * 32;
    float yn = (vals[j] - mu) * rstd * gamma[c] + beta[c];
    float z = zx[row * NPAD + c];
    ybf[row * DIN + c] = __float2bfloat16(yn * z);
  }
}

// ---------------- out-proj GEMM + residual, scatter to (B,C,H,W) ----------------
__global__ __launch_bounds__(256) void k_gemm_out(const __hip_bfloat16* __restrict__ A,
                                                  const __hip_bfloat16* __restrict__ W,
                                                  const float* __restrict__ u,
                                                  float* __restrict__ outp) {
  int wave = threadIdx.x >> 5;
  int tile = blockIdx.x * 8 + wave;          // 196 * 6 tiles of 32x32
  if (tile >= 196 * 6) return;
  int mt = tile / 6, nt = tile % 6;
  int m0 = mt * 32, n0 = nt * 32;
  v8f a00 = vz(), a01 = vz(), a10 = vz(), a11 = vz();
  for (int k0 = 0; k0 < DIN; k0 += 32) {
    v16bf fa0 = ldfragA(A + (size_t)m0 * DIN + k0, DIN);
    v16bf fa1 = ldfragA(A + (size_t)(m0 + 16) * DIN + k0, DIN);
    v16bf fb0 = ldfragB(W + (size_t)n0 * DIN + k0, DIN);
    v16bf fb1 = ldfragB(W + (size_t)(n0 + 16) * DIN + k0, DIN);
    a00 = wmma_bf(fa0, fb0, a00);
    a01 = wmma_bf(fa0, fb1, a01);
    a10 = wmma_bf(fa1, fb0, a10);
    a11 = wmma_bf(fa1, fb1, a11);
  }
  int lane = threadIdx.x & 31;
  v8f accs[4] = {a00, a01, a10, a11};
  int moff[4] = {0, 0, 16, 16};
  int noff[4] = {0, 16, 0, 16};
#pragma unroll
  for (int tsub = 0; tsub < 4; tsub++) {
    int c  = n0 + noff[tsub] + (lane & 15);
    int mB = m0 + moff[tsub] + ((lane >> 4) << 3);
#pragma unroll
    for (int i = 0; i < 8; i++) {
      int row = mB + i; int b = row / L; int l = row % L;
      size_t o = ((size_t)b * DM + c) * L + l;
      outp[o] = accs[tsub][i] + u[o];
    }
  }
}

// ---------------- host launcher ----------------
extern "C" void kernel_launch(void* const* d_in, const int* in_sizes, int n_in,
                              void* d_out, int out_size, void* d_ws, size_t ws_size,
                              hipStream_t stream) {
  const float* u        = (const float*)d_in[0];
  const float* state_in = (const float*)d_in[1];
  const float* W_in     = (const float*)d_in[2];
  const float* dt_bias  = (const float*)d_in[3];
  const float* conv_w   = (const float*)d_in[4];
  const float* conv_b   = (const float*)d_in[5];
  const float* A_log    = (const float*)d_in[6];
  const float* ln_gamma = (const float*)d_in[7];
  const float* ln_beta  = (const float*)d_in[8];
  const float* W_out    = (const float*)d_in[9];
  float* outp = (float*)d_out;
  (void)in_sizes; (void)n_in; (void)out_size; (void)ws_size;

  char* ws = (char*)d_ws;
  size_t off = 0;
  auto take = [&](size_t bytes) -> char* {
    char* p = ws + off;
    off += (bytes + 255) & ~(size_t)255;
    return p;
  };
  __hip_bfloat16* ubf    = (__hip_bfloat16*)take((size_t)M * DM * 2);
  __hip_bfloat16* winbf  = (__hip_bfloat16*)take((size_t)NPAD * DM * 2);
  __hip_bfloat16* woutbf = (__hip_bfloat16*)take((size_t)DM * DIN * 2);
  float* zx   = (float*)take((size_t)M * NPAD * 4);
  float* dtv  = (float*)take((size_t)M * NH * 4);
  float* xbcc = (float*)take((size_t)M * CONVD * 4);
  float* dcum = (float*)take((size_t)BB * NH * L * 4);
  __hip_bfloat16* cbf  = (__hip_bfloat16*)take((size_t)M * HD * 2);
  __hip_bfloat16* bbf  = (__hip_bfloat16*)take((size_t)M * HD * 2);
  __hip_bfloat16* xtbf = (__hip_bfloat16*)take((size_t)BB * NH * HD * L * 2);
  float* Y = (float*)take((size_t)M * DIN * 4);
  __hip_bfloat16* ybf = (__hip_bfloat16*)take((size_t)M * DIN * 2);

  float* fstate_out = outp + (size_t)BB * DM * L;   // tuple tail: final_state

  k_u2bf   <<<(M * DM + 255) / 256, 256, 0, stream>>>(u, ubf);
  k_win2bf <<<(NPAD * DM + 255) / 256, 256, 0, stream>>>(W_in, winbf);
  k_wout2bf<<<(DM * DIN + 255) / 256, 256, 0, stream>>>(W_out, woutbf);
  k_gemm_in<<<(196 * 29 + 7) / 8, 256, 0, stream>>>(ubf, winbf, zx);
  k_dt     <<<(M * NH + 255) / 256, 256, 0, stream>>>(zx, dt_bias, dtv);
  k_conv   <<<(M * CONVD + 255) / 256, 256, 0, stream>>>(zx, conv_w, conv_b, xbcc);
  k_cumsum <<<BB * NH, 256, 0, stream>>>(dtv, A_log, dcum);
  k_prep   <<<(M * CONVD + 255) / 256, 256, 0, stream>>>(xbcc, dtv, xtbf, bbf, cbf);
  k_attn   <<<dim3(L / 64, BB * NH), 128, 0, stream>>>(cbf, bbf, xtbf, dcum, Y);
  k_states <<<BB * NH, 256, 0, stream>>>(bbf, xtbf, dcum, state_in, fstate_out);
  k_yoff   <<<(M * NH * HD + 255) / 256, 256, 0, stream>>>(cbf, state_in, dcum, Y);
  k_ln     <<<M / 8, 256, 0, stream>>>(Y, zx, ln_gamma, ln_beta, ybf);
  k_gemm_out<<<(196 * 6 + 7) / 8, 256, 0, stream>>>(ybf, woutbf, u, outp);
}
